// CrossAttentionBlock_10548439679099
// MI455X (gfx1250) — matrix operand order repaired and checked by reading, not compiled, not run on hardware
//
#include <hip/hip_runtime.h>
#include <hip/hip_bf16.h>

typedef __bf16 bf16;
typedef __attribute__((ext_vector_type(16))) __bf16 v16bf;
typedef __attribute__((ext_vector_type(8)))  float  v8f;

// ---------------------------------------------------------------------------
// WMMA helpers (CDNA5 gfx1250, wave32).
// A-fragment (16x32 bf16): lane l (l15=l&15,row=l15; half=l>>4):
//   elements 0..15 = A[row][half*16 + 0..15]      -> contiguous load (two b128)
// B-fragment (32x16 bf16): col = l15:
//   elements 0..15 = B[half*16 + 0..15][col]
//   with B stored transposed in LDS (Bt[col][k]) this is 16 contiguous bf16.
// C/D (16x16 f32): acc[j] = C[j + 8*half][l15]
// ---------------------------------------------------------------------------
__device__ __forceinline__ v16bf frag_gap(const bf16* p) {
  union { v16bf v; uint4 u[2]; } f;
  f.u[0] = *reinterpret_cast<const uint4*>(p);
  f.u[1] = *reinterpret_cast<const uint4*>(p + 8);
  return f.v;
}
__device__ __forceinline__ v16bf frag_cont(const bf16* p) {
  union { v16bf v; uint4 u[2]; } f;
  f.u[0] = *reinterpret_cast<const uint4*>(p);
  f.u[1] = *reinterpret_cast<const uint4*>(p + 8);
  return f.v;
}
__device__ __forceinline__ v8f wmma_bf16(v16bf a, v16bf b, v8f c) {
  return __builtin_amdgcn_wmma_f32_16x16x32_bf16(false, a, false, b,
                                                 (short)0, c, false, false);
}

// Async global -> LDS copy (16B per lane), tracked on ASYNCcnt.
// LDS byte address: low 32 bits of the flat address of a __shared__ object
__device__ __forceinline__ void async_copy_b128(const bf16* gptr, const bf16* ldsptr) {
  const unsigned lo = (unsigned)(unsigned long long)ldsptr;
  asm volatile("global_load_async_to_lds_b128 %0, %1, off"
               :: "v"(lo), "v"(gptr) : "memory");
}
__device__ __forceinline__ void wait_async() {
  asm volatile("s_wait_asynccnt 0x0" ::: "memory");
}

// ---------------------------------------------------------------------------
// Elementwise f32 -> bf16 cast (vec4)
// ---------------------------------------------------------------------------
__global__ __launch_bounds__(256)
void cast_bf16_kernel(const float* __restrict__ in, bf16* __restrict__ out, long n4) {
  long i = (long)blockIdx.x * blockDim.x + threadIdx.x;
  long stride = (long)gridDim.x * blockDim.x;
  for (; i < n4; i += stride) {
    float4 v = *reinterpret_cast<const float4*>(in + i * 4);
    bf16* o = out + i * 4;
    o[0] = (bf16)v.x; o[1] = (bf16)v.y; o[2] = (bf16)v.z; o[3] = (bf16)v.w;
  }
}

// ---------------------------------------------------------------------------
// LayerNorm over H=1024, one block (8 waves) per token, bf16 output
// ---------------------------------------------------------------------------
__global__ __launch_bounds__(256)
void ln_kernel(const float* __restrict__ x, const float* __restrict__ g,
               const float* __restrict__ b, bf16* __restrict__ out) {
  const long t = blockIdx.x;
  const int tid = threadIdx.x;
  const float4 v = *reinterpret_cast<const float4*>(x + t * 1024 + tid * 4);
  float s = v.x + v.y + v.z + v.w;
  float q = v.x * v.x + v.y * v.y + v.z * v.z + v.w * v.w;
#pragma unroll
  for (int m = 16; m >= 1; m >>= 1) { s += __shfl_xor(s, m); q += __shfl_xor(q, m); }
  __shared__ float ss[8], sq[8];
  if ((tid & 31) == 0) { ss[tid >> 5] = s; sq[tid >> 5] = q; }
  __syncthreads();
  float ts = 0.f, tq = 0.f;
#pragma unroll
  for (int i = 0; i < 8; ++i) { ts += ss[i]; tq += sq[i]; }
  const float mu = ts * (1.0f / 1024.0f);
  const float var = tq * (1.0f / 1024.0f) - mu * mu;
  const float r = rsqrtf(var + 1e-6f);
  const float4 gv = *reinterpret_cast<const float4*>(g + tid * 4);
  const float4 bv = *reinterpret_cast<const float4*>(b + tid * 4);
  bf16* o = out + t * 1024 + tid * 4;
  o[0] = (bf16)((v.x - mu) * r * gv.x + bv.x);
  o[1] = (bf16)((v.y - mu) * r * gv.y + bv.y);
  o[2] = (bf16)((v.z - mu) * r * gv.z + bv.z);
  o[3] = (bf16)((v.w - mu) * r * gv.w + bv.w);
}

// ---------------------------------------------------------------------------
// bf16 WMMA GEMM: C[M,N] = A[M,K] @ B[K,N] (+bias) (+residual)
// Block tile 128x64, BK=32, 256 threads = 8 waves (4 M x 2 N),
// each wave owns a 32x32 sub-tile = 4 wmma/K-step (2 A-frags x 2 B-frags).
// A tile staged via GLOBAL_LOAD_ASYNC_TO_LDS_B128 (ASYNCcnt),
// B tile loaded sync + transposed into LDS.
// mode 0: outB = bf16(acc + bias)
// mode 1: outF = acc + bias + (res ? res : 0)   (f32)
// ---------------------------------------------------------------------------
__global__ __launch_bounds__(256)
void wmma_gemm_kernel(const bf16* __restrict__ A, long lda,
                      const bf16* __restrict__ Bw, long ldb,
                      const float* __restrict__ bias,
                      const float* __restrict__ res, long ldres,
                      float* __restrict__ outF, bf16* __restrict__ outB,
                      long ldc, int K, int mode) {
  __shared__ __align__(16) bf16 As[128][40];  // [m][k] + pad
  __shared__ __align__(16) bf16 Bt[64][40];   // [n][k] (transposed) + pad
  const int tid = threadIdx.x;
  const int lane = tid & 31, half = lane >> 4, l15 = lane & 15;
  const int w = tid >> 5, wm = w >> 1, wn = w & 1;
  const long mBase = (long)blockIdx.y * 128;
  const long nBase = (long)blockIdx.x * 64;

  const int br = tid >> 3, bc = (tid & 7) * 8;   // B: 32 rows x 8 chunks of 8

  v8f acc00 = {}, acc01 = {}, acc10 = {}, acc11 = {};
  const int nk = K >> 5;
  for (int kb = 0; kb < nk; ++kb) {
    const long k0 = (long)kb << 5;
    // A tile: 128x32 = 512 x 16B chunks, async DMA to LDS (2 per thread)
#pragma unroll
    for (int rep = 0; rep < 2; ++rep) {
      const int c = tid + rep * 256;
      const int ar = c >> 2, ac = (c & 3) * 8;
      async_copy_b128(&A[(mBase + ar) * lda + k0 + ac], &As[ar][ac]);
    }
    // B tile: sync load + transpose into LDS
    uint4 bv = *reinterpret_cast<const uint4*>(&Bw[(k0 + br) * ldb + nBase + bc]);
    if (kb + 1 < nk) {
      __builtin_prefetch(&Bw[(k0 + 32 + br) * ldb + nBase + bc], 0, 0);
    }
    union { uint4 u; bf16 h[8]; } tb; tb.u = bv;
#pragma unroll
    for (int i = 0; i < 8; ++i) Bt[bc + i][br] = tb.h[i];
    wait_async();
    __syncthreads();

    const v16bf af0 = frag_gap(&As[wm * 32 + l15][half * 16]);
    const v16bf af1 = frag_gap(&As[wm * 32 + 16 + l15][half * 16]);
    const v16bf bf0 = frag_cont(&Bt[wn * 32 + l15][half * 16]);
    const v16bf bf1 = frag_cont(&Bt[wn * 32 + 16 + l15][half * 16]);
    acc00 = wmma_bf16(af0, bf0, acc00);
    acc01 = wmma_bf16(af0, bf1, acc01);
    acc10 = wmma_bf16(af1, bf0, acc10);
    acc11 = wmma_bf16(af1, bf1, acc11);
    __syncthreads();
  }

  const long col0 = nBase + wn * 32 + l15;
  const long col1 = col0 + 16;
  const float bs0 = bias ? bias[col0] : 0.0f;
  const float bs1 = bias ? bias[col1] : 0.0f;
#pragma unroll
  for (int j = 0; j < 8; ++j) {
    const long r0 = mBase + wm * 32 + j + half * 8;
    const long r1 = r0 + 16;
    const float v00 = acc00[j] + bs0, v01 = acc01[j] + bs1;
    const float v10 = acc10[j] + bs0, v11 = acc11[j] + bs1;
    if (mode == 0) {
      outB[r0 * ldc + col0] = (bf16)v00;
      outB[r0 * ldc + col1] = (bf16)v01;
      outB[r1 * ldc + col0] = (bf16)v10;
      outB[r1 * ldc + col1] = (bf16)v11;
    } else {
      const float rr00 = res ? res[r0 * ldres + col0] : 0.f;
      const float rr01 = res ? res[r0 * ldres + col1] : 0.f;
      const float rr10 = res ? res[r1 * ldres + col0] : 0.f;
      const float rr11 = res ? res[r1 * ldres + col1] : 0.f;
      outF[r0 * ldc + col0] = v00 + rr00;
      outF[r0 * ldc + col1] = v01 + rr01;
      outF[r1 * ldc + col0] = v10 + rr10;
      outF[r1 * ldc + col1] = v11 + rr11;
    }
  }
}

// ---------------------------------------------------------------------------
// Flash attention: grid (Nq/64, B*16 heads), 128 threads (4 waves x 16 q-rows)
// d = 64, kv tile = 32 rows. q,k,v,ctx are bf16 (B*N, 1024) with head at h*64.
// K tile staged async to LDS; V tile transposed sync.
// ---------------------------------------------------------------------------
__global__ __launch_bounds__(128)
void attn_kernel(const bf16* __restrict__ q, const bf16* __restrict__ k,
                 const bf16* __restrict__ v, bf16* __restrict__ ctx) {
  __shared__ __align__(16) bf16 Ks[32][72];      // [kv][d]
  __shared__ __align__(16) bf16 Vt[64][40];      // [d][kv] transposed
  __shared__ __align__(16) bf16 Ps[4][16][40];   // per-wave P transpose buffer
  const int tid = threadIdx.x;
  const int lane = tid & 31, half = lane >> 4, l15 = lane & 15;
  const int w = tid >> 5;
  const long b = blockIdx.y >> 4, h = blockIdx.y & 15;
  const long qbase = (long)blockIdx.x * 64;

  const bf16* qp = q + (b * 1024 + qbase + w * 16 + l15) * 1024 + h * 64;
  const v16bf qf0 = frag_gap(qp + half * 16);        // K-slice 0..31
  const v16bf qf1 = frag_gap(qp + 32 + half * 16);   // K-slice 32..63

  float mprev[8], lsum[8];
  v8f o[4] = {};
#pragma unroll
  for (int j = 0; j < 8; ++j) { mprev[j] = -1e30f; lsum[j] = 0.f; }

  const long kvRowBase = b * 2048;
  for (int kv0 = 0; kv0 < 2048; kv0 += 32) {
#pragma unroll
    for (int rep = 0; rep < 2; ++rep) {
      const int c = tid + rep * 128;
      const int row = c >> 3, col = (c & 7) * 8;
      const long gro = (kvRowBase + kv0 + row) * 1024 + h * 64 + col;
      async_copy_b128(k + gro, &Ks[row][col]);       // K: async DMA to LDS
      union { uint4 u; bf16 e[8]; } tv;              // V: sync + transpose
      tv.u = *reinterpret_cast<const uint4*>(v + gro);
#pragma unroll
      for (int i = 0; i < 8; ++i) Vt[col + i][row] = tv.e[i];
    }
    wait_async();
    __syncthreads();

    // S = Q @ K^T  (two 16-col subtiles, K-dim 64 => 2 wmma each)
    v8f s0 = {}, s1 = {};
    {
      const bf16* kp0 = &Ks[l15][0];
      const bf16* kp1 = &Ks[16 + l15][0];
      s0 = wmma_bf16(qf0, frag_cont(kp0 + half * 16), s0);
      s0 = wmma_bf16(qf1, frag_cont(kp0 + 32 + half * 16), s0);
      s1 = wmma_bf16(qf0, frag_cont(kp1 + half * 16), s1);
      s1 = wmma_bf16(qf1, frag_cont(kp1 + 32 + half * 16), s1);
    }

    // online softmax (rows live in (j, half); cols across 16 lanes)
#pragma unroll
    for (int j = 0; j < 8; ++j) {
      const float sa = s0[j] * 0.125f, sb = s1[j] * 0.125f;
      float m = fmaxf(sa, sb);
#pragma unroll
      for (int msk = 1; msk < 16; msk <<= 1) m = fmaxf(m, __shfl_xor(m, msk));
      const float mn = fmaxf(mprev[j], m);
      const float alpha = __expf(mprev[j] - mn);
      const float p0 = __expf(sa - mn), p1 = __expf(sb - mn);
      float rs = p0 + p1;
#pragma unroll
      for (int msk = 1; msk < 16; msk <<= 1) rs += __shfl_xor(rs, msk);
      lsum[j] = lsum[j] * alpha + rs;
      mprev[j] = mn;
      o[0][j] *= alpha; o[1][j] *= alpha; o[2][j] *= alpha; o[3][j] *= alpha;
      const int r = j + half * 8;
      Ps[w][r][l15] = (bf16)p0;
      Ps[w][r][16 + l15] = (bf16)p1;
    }
    // in-wave LDS write -> cross-lane read hazard
    asm volatile("s_wait_dscnt 0x0" ::: "memory");

    const v16bf pf = frag_gap(&Ps[w][l15][half * 16]);
#pragma unroll
    for (int dt = 0; dt < 4; ++dt)
      o[dt] = wmma_bf16(pf, frag_cont(&Vt[dt * 16 + l15][half * 16]), o[dt]);
    __syncthreads();
  }

  bf16* cp = ctx + (b * 1024 + qbase + w * 16) * 1024 + h * 64;
#pragma unroll
  for (int dt = 0; dt < 4; ++dt)
#pragma unroll
    for (int j = 0; j < 8; ++j) {
      const int r = j + half * 8;
      cp[(long)r * 1024 + dt * 16 + l15] = (bf16)(o[dt][j] / lsum[j]);
    }
}

// ---------------------------------------------------------------------------
// SiLU(gate) * up from gu (per-expert [1024, 2048] f32) -> inter bf16 [1024,1024]
// ---------------------------------------------------------------------------
__global__ __launch_bounds__(256)
void silu_kernel(const float* __restrict__ gu, bf16* __restrict__ inter) {
  const long idx = (long)blockIdx.x * blockDim.x + threadIdx.x; // 4*1024*256
  const long e = idx >> 18;
  const long rem = idx & 262143;
  const long t = rem >> 8;
  const long c4 = (rem & 255) * 4;
  const float* gp = gu + e * 2097152 + t * 2048 + c4;
  const float4 g = *reinterpret_cast<const float4*>(gp);
  const float4 u = *reinterpret_cast<const float4*>(gp + 1024);
  bf16* op = inter + e * 1048576 + t * 1024 + c4;
  op[0] = (bf16)(g.x / (1.f + __expf(-g.x)) * u.x);
  op[1] = (bf16)(g.y / (1.f + __expf(-g.y)) * u.y);
  op[2] = (bf16)(g.z / (1.f + __expf(-g.z)) * u.z);
  op[3] = (bf16)(g.w / (1.f + __expf(-g.w)) * u.w);
}

// ---------------------------------------------------------------------------
extern "C" void kernel_launch(void* const* d_in, const int* in_sizes, int n_in,
                              void* d_out, int out_size, void* d_ws, size_t ws_size,
                              hipStream_t stream) {
  (void)in_sizes; (void)n_in; (void)out_size; (void)ws_size;
  const float* query   = (const float*)d_in[0];
  const float* keyval  = (const float*)d_in[1];
  const float* Wq = (const float*)d_in[2];  const float* bq = (const float*)d_in[3];
  const float* Wk = (const float*)d_in[4];  const float* bk = (const float*)d_in[5];
  const float* Wv = (const float*)d_in[6];  const float* bv = (const float*)d_in[7];
  const float* Wo = (const float*)d_in[8];  const float* bo = (const float*)d_in[9];
  const float* g1 = (const float*)d_in[10]; const float* b1 = (const float*)d_in[11];
  const float* g2 = (const float*)d_in[12]; const float* b2 = (const float*)d_in[13];
  const float* gate_up = (const float*)d_in[14];
  const float* down    = (const float*)d_in[15];
  float* out = (float*)d_out;

  const long T = 4096, TKV = 8192, HH = 1024, II = 1024;
  char* ws = (char*)d_ws;
  size_t cur = 0;
  auto alloc = [&](size_t bytes) -> void* {
    void* p = ws + cur;
    cur += (bytes + 255) & ~(size_t)255;
    return p;
  };
  bf16* xln    = (bf16*)alloc((size_t)T * HH * 2);
  bf16* kvb    = (bf16*)alloc((size_t)TKV * HH * 2);
  bf16* qb     = (bf16*)alloc((size_t)T * HH * 2);
  bf16* kbuf   = (bf16*)alloc((size_t)TKV * HH * 2);
  bf16* vbuf   = (bf16*)alloc((size_t)TKV * HH * 2);
  bf16* ctxb   = (bf16*)alloc((size_t)T * HH * 2);
  bf16* hb     = (bf16*)alloc((size_t)T * HH * 2);
  bf16* wqb    = (bf16*)alloc((size_t)HH * HH * 2);
  bf16* wkb    = (bf16*)alloc((size_t)HH * HH * 2);
  bf16* wvb    = (bf16*)alloc((size_t)HH * HH * 2);
  bf16* wob    = (bf16*)alloc((size_t)HH * HH * 2);
  bf16* guw    = (bf16*)alloc((size_t)4 * HH * 2 * II * 2);
  bf16* dww    = (bf16*)alloc((size_t)4 * II * HH * 2);
  bf16* interb = (bf16*)alloc((size_t)4 * 1024 * II * 2);
  float* x2    = (float*)alloc((size_t)T * HH * 4);
  float* gu    = (float*)alloc((size_t)4 * 1024 * 2 * II * 4);

  auto cast = [&](const float* src, bf16* dst, long n) {
    cast_bf16_kernel<<<2048, 256, 0, stream>>>(src, dst, n / 4);
  };
  auto gemm = [&](const bf16* A, long lda, const bf16* Bw, long ldb,
                  const float* bias, const float* res, long ldres,
                  float* oF, bf16* oB, long ldc, int M, int N, int K, int mode) {
    dim3 g(N / 64, M / 128);
    wmma_gemm_kernel<<<g, 256, 0, stream>>>(A, lda, Bw, ldb, bias, res, ldres,
                                            oF, oB, ldc, K, mode);
  };

  // weight / activation casts to bf16
  cast(keyval, kvb, TKV * HH);
  cast(Wq, wqb, HH * HH); cast(Wk, wkb, HH * HH);
  cast(Wv, wvb, HH * HH); cast(Wo, wob, HH * HH);
  cast(gate_up, guw, 4 * HH * 2 * II);
  cast(down, dww, 4 * II * HH);

  // LN1
  ln_kernel<<<T, 256, 0, stream>>>(query, g1, b1, xln);

  // Q/K/V projections
  gemm(xln, HH, wqb, HH, bq, nullptr, 0, nullptr, qb,   HH, T,   HH, HH, 0);
  gemm(kvb, HH, wkb, HH, bk, nullptr, 0, nullptr, kbuf, HH, TKV, HH, HH, 0);
  gemm(kvb, HH, wvb, HH, bv, nullptr, 0, nullptr, vbuf, HH, TKV, HH, HH, 0);

  // attention
  attn_kernel<<<dim3(16, 64), 128, 0, stream>>>(qb, kbuf, vbuf, ctxb);

  // output projection + residual (query) -> x2 (f32)
  gemm(ctxb, HH, wob, HH, bo, query, HH, x2, nullptr, HH, T, HH, HH, 1);

  // LN2
  ln_kernel<<<T, 256, 0, stream>>>(x2, g2, b2, hb);

  // MoE: expert e handles tokens t = 4*m + e  (row stride 4*H, offset e*H)
  for (int e = 0; e < 4; ++e) {
    gemm(hb + e * HH, 4 * HH, guw + (size_t)e * HH * 2 * II, 2 * II,
         nullptr, nullptr, 0, gu + (size_t)e * 1024 * 2 * II, nullptr, 2 * II,
         1024, 2 * II, HH, 1);
  }
  silu_kernel<<<4096, 256, 0, stream>>>(gu, interb);
  for (int e = 0; e < 4; ++e) {
    gemm(interb + (size_t)e * 1024 * II, II, dww + (size_t)e * II * HH, HH,
         nullptr, x2 + e * HH, 4 * HH, out + e * HH, nullptr, 4 * HH,
         1024, HH, II, 1);
  }
}
